// SimplifiedSSM_30236569764383
// MI455X (gfx1250) — matrix operand order repaired and checked by reading, not compiled, not run on hardware
//
#include <hip/hip_runtime.h>
#include <hip/hip_bf16.h>
#include <math.h>

// Problem constants (from reference)
#define D_MODEL 1024
#define D_STATE 128
#define BATCH   8
#define SEQ     4096
#define M_TOT   (BATCH * SEQ)   // 32768 rows

typedef __attribute__((ext_vector_type(16))) _Float16 v16h;
typedef __attribute__((ext_vector_type(8)))  _Float16 v8h;
typedef __attribute__((ext_vector_type(2)))  _Float16 v2h;
typedef __attribute__((ext_vector_type(8)))  float    v8f;
typedef __attribute__((ext_vector_type(4)))  int      v4i;

#define SHUF16(lo8, hi8) __builtin_shufflevector(lo8, hi8, \
            0,1,2,3,4,5,6,7,8,9,10,11,12,13,14,15)

#if defined(__gfx1250__) && __has_builtin(__builtin_amdgcn_global_load_async_to_lds_b128)
#define HAVE_ASYNC_LDS 1
typedef __attribute__((address_space(1))) v4i v4i_global;  // global-AS int4
typedef __attribute__((address_space(3))) v4i v4i_lds;     // LDS-AS int4
#endif

// ---------------------------------------------------------------------------
// Kernel 1: xB[m][n] = sum_k x[m][k] * Bw[k][n]
//   M = 32768, K = 1024, N = 128.  f16 WMMA, f32 accumulate.
//   Block: 256 threads = 8 waves, M tile = 64 rows (4 subtiles of 16 per
//   wave sharing one B fragment), N = 128 (wave w owns columns [16w,16w+16)).
//   K staged in steps of 32 through LDS with packed v2h stores.
// ---------------------------------------------------------------------------
__global__ __launch_bounds__(256)
void xb_gemm_kernel(const float* __restrict__ x,
                    const float* __restrict__ Bw,
                    float* __restrict__ xb)
{
    __shared__ __align__(16) _Float16 Xs[64][32];    // 4 KB
    __shared__ __align__(16) _Float16 Ws[128][32];   // 8 KB (N-major)

    const int tid  = threadIdx.x;
    const int wv   = tid >> 5;
    const int lane = tid & 31;
    const int ln16 = lane & 15;
    const int hi   = lane >> 4;
    const int m0   = blockIdx.x * 64;

    v8f acc[4] = {};

    for (int kk = 0; kk < D_MODEL / 32; ++kk) {
        const int K0 = kk * 32;

        // Stage X tile 64x32: 1024 f16-pairs, 4 per thread, packed b32 stores
        #pragma unroll
        for (int j = 0; j < 4; ++j) {
            int p = tid + 256 * j;
            int r = p >> 4, kp = p & 15;
            float2 xv = *(const float2*)&x[(size_t)(m0 + r) * D_MODEL + K0 + 2 * kp];
            v2h h2; h2.x = (_Float16)xv.x; h2.y = (_Float16)xv.y;
            *(v2h*)&Xs[r][2 * kp] = h2;
        }
        // Stage W tile transposed Ws[n][k]: 2048 pairs, 8 per thread
        #pragma unroll
        for (int j = 0; j < 8; ++j) {
            int p = tid + 256 * j;
            int n = p & 127, kp = p >> 7;
            float w0 = Bw[(size_t)(K0 + 2 * kp)     * D_STATE + n];
            float w1 = Bw[(size_t)(K0 + 2 * kp + 1) * D_STATE + n];
            v2h h2; h2.x = (_Float16)w0; h2.y = (_Float16)w1;
            *(v2h*)&Ws[n][2 * kp] = h2;
        }
        // Hint the next K-step of the streaming x rows into cache
        if (K0 + 32 < D_MODEL)
            __builtin_prefetch(&x[(size_t)(m0 + (tid >> 4)) * D_MODEL + K0 + 32], 0, 1);
        __syncthreads();

        // B fragment (32x16): lane -> column 16*wv+(lane&15), K = 8*hi + e (e<8), 16 + 8*hi + (e-8) (e>=8)
        v8h blo = *(const v8h*)&Ws[16 * wv + ln16][8 * hi];
        v8h bhi = *(const v8h*)&Ws[16 * wv + ln16][16 + 8 * hi];
        v16h bv = SHUF16(blo, bhi);

        // 4 A fragments / 4 WMMAs reuse bv
        #pragma unroll
        for (int mi = 0; mi < 4; ++mi) {
            const int r = 16 * mi + ln16;
            v8h alo = *(const v8h*)&Xs[r][8 * hi];
            v8h ahi = *(const v8h*)&Xs[r][16 + 8 * hi];
            v16h av = SHUF16(alo, ahi);
            acc[mi] = __builtin_amdgcn_wmma_f32_16x16x32_f16(
                          false, av, false, bv, (short)0, acc[mi], false, false);
        }
        __syncthreads();
    }

    // C layout: VGPR r -> row 16*mi + r + 8*hi, column 16*wv + (lane&15)
    const int n = 16 * wv + ln16;
    #pragma unroll
    for (int mi = 0; mi < 4; ++mi) {
        #pragma unroll
        for (int r = 0; r < 8; ++r) {
            int m = m0 + 16 * mi + r + 8 * hi;
            xb[(size_t)m * D_STATE + n] = acc[mi][r];
        }
    }
}

// ---------------------------------------------------------------------------
// Kernel 2: sequential recurrence per batch (8 blocks, 128 threads).
//   h_t = tanh(h_{t-1} @ A + xb_t); emit h as f16 for the output GEMM.
//   A (64 KB) cached in LDS; h broadcast through LDS each step.
// ---------------------------------------------------------------------------
__global__ __launch_bounds__(128)
void ssm_scan_kernel(const float* __restrict__ A,
                     const float* __restrict__ xb,
                     _Float16* __restrict__ hf16)
{
    __shared__ float As[D_STATE * D_STATE];  // As[k*128 + t] = A[k][t]
    __shared__ float hs[D_STATE];

    const int t = threadIdx.x;
    const int b = blockIdx.x;

    for (int i = t; i < D_STATE * D_STATE; i += 128)
        As[i] = A[i];
    hs[t] = 0.0f;
    __syncthreads();

    const float* xb_b = xb + (size_t)b * SEQ * D_STATE;
    _Float16*    h_b  = hf16 + (size_t)b * SEQ * D_STATE;

    for (int s = 0; s < SEQ; ++s) {
        float acc = xb_b[(size_t)s * D_STATE + t];
        #pragma unroll 8
        for (int k = 0; k < D_STATE; ++k)
            acc = fmaf(hs[k], As[k * D_STATE + t], acc);
        float hn = tanhf(acc);
        __syncthreads();
        hs[t] = hn;
        h_b[(size_t)s * D_STATE + t] = (_Float16)hn;
        __syncthreads();
    }
}

// ---------------------------------------------------------------------------
// Kernel 3: y[m][d] = sum_k h[m][k] * Cm[d][k] + Dv[d]
//   M = 32768, K = 128, N = 1024. f16 WMMA, f32 accumulate.
//   Grid (512, 8): x -> 64-row M tile, y -> 128-wide D chunk.
//   H tile staged via async global->LDS copy (raw f16, no conversion).
// ---------------------------------------------------------------------------
__global__ __launch_bounds__(256)
void y_gemm_kernel(const _Float16* __restrict__ hf16,
                   const float* __restrict__ Cm,
                   const float* __restrict__ Dv,
                   float* __restrict__ y)
{
    __shared__ __align__(16) _Float16 Hs[64][D_STATE];    // 16 KB
    __shared__ __align__(16) _Float16 Cs[128][D_STATE];   // 32 KB

    const int tid  = threadIdx.x;
    const int wv   = tid >> 5;
    const int lane = tid & 31;
    const int ln16 = lane & 15;
    const int hi   = lane >> 4;
    const int m0   = blockIdx.x * 64;
    const int d0   = blockIdx.y * 128;

    // Stage H tile rows 0..31 only (32x128 f16 = 8 KB raw copy): 2 x 16B per thread
    {
        _Float16*       hflat = &Hs[0][0];
        const _Float16* gsrc  = hf16 + (size_t)m0 * D_STATE;
        #pragma unroll
        for (int j = 0; j < 2; ++j) {
            int off8 = 8 * (tid + 256 * j);       // in f16 elements
#ifdef HAVE_ASYNC_LDS
            __builtin_amdgcn_global_load_async_to_lds_b128(
                (v4i_global*)(uintptr_t)(gsrc + off8),
                (v4i_lds*)(uint32_t)(uintptr_t)(hflat + off8), 0, 0);
#else
            *(v8h*)(hflat + off8) = *(const v8h*)(gsrc + off8);
#endif
        }
    }
    // Stage C chunk: Cs[dd][k] = (f16)Cm[d0+dd][k], packed pair stores
    {
        _Float16* cflat = &Cs[0][0];
        #pragma unroll
        for (int j = 0; j < 32; ++j) {
            int p = tid + 256 * j;                // pair index, 8192 pairs
            float2 cv = *(const float2*)&Cm[(size_t)d0 * D_STATE + 2 * p];
            v2h h2; h2.x = (_Float16)cv.x; h2.y = (_Float16)cv.y;
            *(v2h*)(cflat + 2 * p) = h2;
        }
    }
#ifdef HAVE_ASYNC_LDS
#if __has_builtin(__builtin_amdgcn_s_wait_asynccnt)
    __builtin_amdgcn_s_wait_asynccnt(0);
#else
    asm volatile("s_wait_asynccnt 0" ::: "memory");
#endif
#endif
    __syncthreads();

    v8f acc[4] = {};
    #pragma unroll
    for (int kk = 0; kk < D_STATE / 32; ++kk) {
        const int K0 = kk * 32;

        v8h blo = *(const v8h*)&Cs[16 * wv + ln16][K0 + 8 * hi];
        v8h bhi = *(const v8h*)&Cs[16 * wv + ln16][K0 + 16 + 8 * hi];
        v16h bv = SHUF16(blo, bhi);

        #pragma unroll
        for (int mi = 0; mi < 4; ++mi) {
            const int r = 16 * mi + ln16;
            v8h alo = *(const v8h*)&Hs[r][K0 + 8 * hi];
            v8h ahi = *(const v8h*)&Hs[r][K0 + 16 + 8 * hi];
            v16h av = SHUF16(alo, ahi);
            acc[mi] = __builtin_amdgcn_wmma_f32_16x16x32_f16(
                          false, av, false, bv, (short)0, acc[mi], false, false);
        }
    }

    const int d     = d0 + 16 * wv + ln16;
    const float bias = Dv[d];
    #pragma unroll
    for (int mi = 0; mi < 4; ++mi) {
        #pragma unroll
        for (int r = 0; r < 8; ++r) {
            int m = m0 + 16 * mi + r + 8 * hi;
            y[(size_t)m * D_MODEL + d] = acc[mi][r] + bias;
        }
    }
}

// ---------------------------------------------------------------------------
extern "C" void kernel_launch(void* const* d_in, const int* in_sizes, int n_in,
                              void* d_out, int out_size, void* d_ws, size_t ws_size,
                              hipStream_t stream)
{
    (void)in_sizes; (void)n_in; (void)out_size; (void)ws_size;
    const float* x  = (const float*)d_in[0];  // [8,4096,1024]
    const float* A  = (const float*)d_in[1];  // [128,128]
    const float* Bw = (const float*)d_in[2];  // [1024,128]
    const float* Cm = (const float*)d_in[3];  // [1024,128]
    const float* Dv = (const float*)d_in[4];  // [1024]
    float* y = (float*)d_out;                 // [8,4096,1024]

    // Workspace: xB f32 (16 MB) then h f16 (8 MB)
    float*    xb = (float*)d_ws;
    _Float16* hf = (_Float16*)((char*)d_ws + (size_t)M_TOT * D_STATE * sizeof(float));

    xb_gemm_kernel<<<dim3(M_TOT / 64), dim3(256), 0, stream>>>(x, Bw, xb);
    ssm_scan_kernel<<<dim3(BATCH), dim3(128), 0, stream>>>(A, xb, hf);
    y_gemm_kernel<<<dim3(M_TOT / 64, D_MODEL / 128), dim3(256), 0, stream>>>(hf, Cm, Dv, y);
}
